// Cross_MultiAttention_446676598865
// MI455X (gfx1250) — matrix operand order repaired and checked by reading, not compiled, not run on hardware
//
#include <hip/hip_runtime.h>
#include <cstdint>
#include <cstddef>

#define BATCH   4u
#define LQ      1024u
#define LK      1024u
#define QDIM    512u
#define EMB     1024u
#define HEADS   16u
#define DEPTH   64u
#define NH      (BATCH*HEADS)
#define SCALE   0.03125f   // EMB^-0.5

typedef __attribute__((ext_vector_type(16))) __bf16 v16bf;
typedef __attribute__((ext_vector_type(8)))  float  v8f;

union Frag { v16bf v; uint32_t u[8]; uint4 q[2]; };

__device__ __forceinline__ uint32_t f2bf(float f) {
  uint32_t u = __float_as_uint(f);
  return (u + 0x7FFFu + ((u >> 16) & 1u)) >> 16;   // RNE
}
__device__ __forceinline__ uint32_t pack_bf16(float a, float b) {
  return f2bf(a) | (f2bf(b) << 16);
}

// ---------------------------------------------------------------------------
// One-shot f32 -> bf16 conversion (each element converted exactly once).
// ---------------------------------------------------------------------------
__global__ __launch_bounds__(256)
void cvt_bf16_kernel(const float* __restrict__ in, uint16_t* __restrict__ out,
                     uint32_t n4)   // n/4 float4 groups
{
  uint32_t i = blockIdx.x * 256u + threadIdx.x;
  const uint32_t stride = gridDim.x * 256u;
  for (; i < n4; i += stride) {
    const float4 f = ((const float4*)in)[i];
    uint2 o;
    o.x = pack_bf16(f.x, f.y);
    o.y = pack_bf16(f.z, f.w);
    ((uint2*)out)[i] = o;
  }
}

// ---------------------------------------------------------------------------
// Projection: dst(bf16) = src[M x 512] @ W[1024 x 512]^T + bias
// src/w pre-converted to bf16: every fragment = 2x global_load_b128.
// transposeV==0 : dst layout [BH][L][DEPTH]   (Q, K)
// transposeV==1 : dst layout [BH][DEPTH][L]   (V, packed 16B stores)
// ---------------------------------------------------------------------------
__global__ __launch_bounds__(128)
void proj_kernel(const uint16_t* __restrict__ src, const uint16_t* __restrict__ w,
                 const float* __restrict__ bias, uint16_t* __restrict__ dst,
                 int transposeV)
{
  const uint32_t lane = threadIdx.x & 31u;
  const uint32_t wave = threadIdx.x >> 5;
  const uint32_t half = lane >> 4;
  const uint32_t l16  = lane & 15u;
  const uint32_t tile_m = blockIdx.y * 64u + wave * 16u;
  const uint32_t tile_n = blockIdx.x * 16u;

  const uint16_t* arow = src + (size_t)(tile_m + l16) * QDIM;  // A row, K-contig
  const uint16_t* brow = w   + (size_t)(tile_n + l16) * QDIM;  // B col, K-contig

  v8f acc = {};
  for (uint32_t k0 = 0; k0 < QDIM; k0 += 32u) {
    __builtin_prefetch(arow + k0 + 128u, 0, 1);   // global_prefetch_b8
    __builtin_prefetch(brow + k0 + 128u, 0, 1);
    Frag a, b;
    a.q[0] = *(const uint4*)(arow + k0 + half * 8u);
    a.q[1] = *(const uint4*)(arow + k0 + 16u + half * 8u);
    b.q[0] = *(const uint4*)(brow + k0 + half * 8u);
    b.q[1] = *(const uint4*)(brow + k0 + 16u + half * 8u);
    acc = __builtin_amdgcn_wmma_f32_16x16x32_bf16(false, a.v, false, b.v,
                                                  (short)0, acc, false, false);
  }
  const float bv = bias[tile_n + l16];

  // C/D layout: vgpr r -> m = tile_m + half*8 + r ; n = tile_n + l16
  const uint32_t b_idx    = tile_m >> 10;              // / LQ
  const uint32_t seq_base = (tile_m & 1023u) + half * 8u;
  const uint32_t h = (tile_n + l16) >> 6;              // / DEPTH
  const uint32_t d = (tile_n + l16) & 63u;             // % DEPTH

  if (!transposeV) {
#pragma unroll
    for (uint32_t r = 0; r < 8u; ++r)
      dst[(((size_t)b_idx * HEADS + h) * LQ + (seq_base + r)) * DEPTH + d] =
          (uint16_t)f2bf(acc[r] + bv);
  } else {
    union { uint16_t s[8]; uint4 q; } pk;
#pragma unroll
    for (uint32_t r = 0; r < 8u; ++r) pk.s[r] = (uint16_t)f2bf(acc[r] + bv);
    *(uint4*)(dst + (((size_t)b_idx * HEADS + h) * DEPTH + d) * LK + seq_base) = pk.q;
  }
}

// ---------------------------------------------------------------------------
// Fused attention: per block = 32 query rows of one (b,h).
// LDS: S f32[32][1024] (128KB) + rinv[32] + Pb bf16[32][1024] (64KB) ~ 192KB.
// ---------------------------------------------------------------------------
__global__ __launch_bounds__(256)
void attn_kernel(const uint16_t* __restrict__ Qh,   // [NH][LQ][DEPTH] bf16
                 const uint16_t* __restrict__ Kh,   // [NH][LK][DEPTH] bf16
                 const uint16_t* __restrict__ Vt,   // [NH][DEPTH][LK] bf16
                 const uint8_t*  __restrict__ mask, // [B][LQ][LK] bool
                 float* __restrict__ att,           // [B][H][LQ][LK]
                 float* __restrict__ outp)          // [B][LQ][EMB]
{
  extern __shared__ float smem[];
  float*    S    = smem;                               // [32][LK] f32
  float*    rinv = smem + 32u * LK;                    // [32]
  uint16_t* Pb   = (uint16_t*)(smem + 32u * LK + 32u); // [32][LK] bf16 (16B aligned)

  const uint32_t tid  = threadIdx.x;
  const uint32_t lane = tid & 31u, wave = tid >> 5;
  const uint32_t half = lane >> 4, l16 = lane & 15u;
  const uint32_t rt = wave >> 2, ct = wave & 3u;   // 2 row-tiles x 4 col-tiles
  const uint32_t bh = blockIdx.y;
  const uint32_t batch = bh >> 4;                  // HEADS = 16
  const uint32_t i0 = blockIdx.x * 32u;

  // Hoisted A fragments: Q rows fixed for the whole block (2x b128 each)
  const uint16_t* qrow = Qh + ((size_t)bh * LQ + (i0 + rt * 16u + l16)) * DEPTH;
  Frag aQ[2];
#pragma unroll
  for (uint32_t d0 = 0; d0 < 2u; ++d0) {
    aQ[d0].q[0] = *(const uint4*)(qrow + d0 * 32u + half * 8u);
    aQ[d0].q[1] = *(const uint4*)(qrow + d0 * 32u + 16u + half * 8u);
  }

  // ---- Pass 1: S = scale(Q K^T) + mask ----
  for (uint32_t j0 = 0; j0 < LK; j0 += 64u) {
    v8f acc = {};
    const uint32_t jj = j0 + ct * 16u + l16;
    const uint16_t* krow = Kh + ((size_t)bh * LK + jj) * DEPTH;
    __builtin_prefetch(Kh + ((size_t)bh * LK + jj + 64u) * DEPTH, 0, 1);
#pragma unroll
    for (uint32_t d0 = 0; d0 < 2u; ++d0) {
      Frag b;
      b.q[0] = *(const uint4*)(krow + d0 * 32u + half * 8u);
      b.q[1] = *(const uint4*)(krow + d0 * 32u + 16u + half * 8u);
      acc = __builtin_amdgcn_wmma_f32_16x16x32_bf16(false, aQ[d0].v, false, b.v,
                                                    (short)0, acc, false, false);
    }
    const uint32_t iL = rt * 16u + half * 8u;
#pragma unroll
    for (uint32_t r = 0; r < 8u; ++r) {
      const uint32_t i = i0 + iL + r;
      float s = acc[r] * SCALE;
      if (mask[((size_t)batch * LQ + i) * LK + jj]) s = -1e9f;
      S[(iL + r) * LK + jj] = s;
    }
  }
  __syncthreads();

  // ---- Pass 2: softmax (8 lanes per row, shfl_xor subgroup reduce) ----
  {
    const uint32_t row = tid >> 3;
    const uint32_t k8  = tid & 7u;
    float* srow = S + row * LK + k8 * 128u;
    float mx = -3.4e38f;
    for (uint32_t t = 0; t < 128u; ++t) mx = fmaxf(mx, srow[t]);
    mx = fmaxf(mx, __shfl_xor(mx, 1, 8));
    mx = fmaxf(mx, __shfl_xor(mx, 2, 8));
    mx = fmaxf(mx, __shfl_xor(mx, 4, 8));
    float sum = 0.f;
    for (uint32_t t = 0; t < 128u; ++t) {
      const float e = __expf(srow[t] - mx);
      srow[t] = e;
      sum += e;
    }
    sum += __shfl_xor(sum, 1, 8);
    sum += __shfl_xor(sum, 2, 8);
    sum += __shfl_xor(sum, 4, 8);
    if (k8 == 0u) rinv[row] = 1.f / sum;
  }
  __syncthreads();

  // normalize: f32 att writeback (float2) + one-shot bf16 pack into Pb
  {
    float* attBlk = att + ((size_t)bh * LQ + i0) * LK;
    for (uint32_t t = 0; t < (32u * LK) / 512u; ++t) {
      const uint32_t idx = (t * 256u + tid) * 2u;    // even: same row for pair
      const float ri = rinv[idx >> 10];
      const float p0 = S[idx] * ri;
      const float p1 = S[idx + 1u] * ri;
      float2 pv; pv.x = p0; pv.y = p1;
      *(float2*)(attBlk + idx) = pv;
      *(uint32_t*)(Pb + idx) = pack_bf16(p0, p1);
    }
  }
  __syncthreads();

  // ---- Pass 3: O = P @ V  (A from bf16 LDS: 2x ds_load_b128) ----
  {
    v8f acc = {};
    for (uint32_t j0 = 0; j0 < LK; j0 += 32u) {
      Frag a, bv;
      const uint16_t* prow = Pb + (rt * 16u + l16) * LK + j0;
      a.q[0] = *(const uint4*)(prow + half * 8u);
      a.q[1] = *(const uint4*)(prow + 16u + half * 8u);
      const uint16_t* vrow = Vt + ((size_t)bh * DEPTH + (ct * 16u + l16)) * LK + j0;
      bv.q[0] = *(const uint4*)(vrow + half * 8u);
      bv.q[1] = *(const uint4*)(vrow + 16u + half * 8u);
      acc = __builtin_amdgcn_wmma_f32_16x16x32_bf16(false, a.v, false, bv.v,
                                                    (short)0, acc, false, false);
    }
    const uint32_t h = bh & 15u;
    const uint32_t dcol = ct * 16u + l16;
#pragma unroll
    for (uint32_t r = 0; r < 8u; ++r) {
      const uint32_t i = i0 + rt * 16u + half * 8u + r;
      outp[((size_t)batch * LQ + i) * EMB + h * DEPTH + dcol] = acc[r];
    }
  }
}

// ---------------------------------------------------------------------------
extern "C" void kernel_launch(void* const* d_in, const int* in_sizes, int n_in,
                              void* d_out, int out_size, void* d_ws, size_t ws_size,
                              hipStream_t stream) {
  (void)in_sizes; (void)n_in; (void)out_size; (void)ws_size;
  const float*   qc   = (const float*)d_in[0];
  const float*   kvc  = (const float*)d_in[1];
  const uint8_t* mask = (const uint8_t*)d_in[2];
  const float*   Wq   = (const float*)d_in[3];
  const float*   bq   = (const float*)d_in[4];
  const float*   Wk   = (const float*)d_in[5];
  const float*   bk   = (const float*)d_in[6];
  const float*   Wv   = (const float*)d_in[7];
  const float*   bv   = (const float*)d_in[8];

  const size_t QKV  = (size_t)NH * LQ * DEPTH;       // 4M bf16 elements each
  const size_t ACT  = (size_t)BATCH * LQ * QDIM;     // 2M elements
  const size_t WSZ  = (size_t)EMB * QDIM;            // 512K elements

  uint16_t* Qh  = (uint16_t*)d_ws;         // 8 MB
  uint16_t* Kh  = Qh  + QKV;               // 8 MB
  uint16_t* Vt  = Kh  + QKV;               // 8 MB
  uint16_t* Qcb = Vt  + QKV;               // 4 MB  bf16(q_candidate)
  uint16_t* Kvb = Qcb + ACT;               // 4 MB  bf16(kv_candidate)
  uint16_t* Wqb = Kvb + ACT;               // 1 MB
  uint16_t* Wkb = Wqb + WSZ;               // 1 MB
  uint16_t* Wvb = Wkb + WSZ;               // 1 MB

  float* outp = (float*)d_out;
  float* att  = outp + (size_t)BATCH * LQ * EMB;

  cvt_bf16_kernel<<<1024, 256, 0, stream>>>(qc,  Qcb, (uint32_t)(ACT / 4u));
  cvt_bf16_kernel<<<1024, 256, 0, stream>>>(kvc, Kvb, (uint32_t)(ACT / 4u));
  cvt_bf16_kernel<<<256,  256, 0, stream>>>(Wq,  Wqb, (uint32_t)(WSZ / 4u));
  cvt_bf16_kernel<<<256,  256, 0, stream>>>(Wk,  Wkb, (uint32_t)(WSZ / 4u));
  cvt_bf16_kernel<<<256,  256, 0, stream>>>(Wv,  Wvb, (uint32_t)(WSZ / 4u));

  dim3 pg(EMB / 16u, (BATCH * LQ) / 64u);
  proj_kernel<<<pg, 128, 0, stream>>>(Qcb, Wqb, bq, Qh, 0);
  proj_kernel<<<pg, 128, 0, stream>>>(Kvb, Wkb, bk, Kh, 0);
  proj_kernel<<<pg, 128, 0, stream>>>(Kvb, Wvb, bv, Vt, 1);

  const size_t smem = (32u * LK + 32u) * sizeof(float)   // S + rinv
                    + (32u * LK) * sizeof(uint16_t);     // Pb  (~192.1 KB)
  attn_kernel<<<dim3(LQ / 32u, NH), 256, smem, stream>>>(Qh, Kh, Vt, mask, att, outp);
}
